// NexToU_Encoder_19344532701694
// MI455X (gfx1250) — matrix operand order repaired and checked by reading, not compiled, not run on hardware
//
#include <hip/hip_runtime.h>

// ---------------------------------------------------------------------------
// NexToU encoder block for MI455X (gfx1250, wave32, WMMA bf16).
// All 1x1x1 convs run as WMMA bf16 GEMMs with fused BN/IN statistics
// (conv bias is dropped: it cancels exactly inside the following BN/IN).
// MRConv aggregation uses the closed form: xj = h - min(residue mins D/H/W).
// ---------------------------------------------------------------------------

typedef __attribute__((ext_vector_type(16))) __bf16 v16bf;
typedef __attribute__((ext_vector_type(8)))  float  v8f;

union U16BF { int4 q[2]; v16bf v; };   // 32B operand: two b128 LDS loads

#define WAVE 32
#define EPSV 1e-5f
#define SLOPEV 0.2f

// ---------------------------------------------------------------------------
// f32 -> bf16 conversion, 4 elements/thread (b128 in, b64 out)
// ---------------------------------------------------------------------------
__global__ __launch_bounds__(256) void cvt_f32_bf16_kernel(
    const float* __restrict__ s, __bf16* __restrict__ d, int n4)
{
    int i = blockIdx.x * 256 + threadIdx.x;
    if (i < n4) {
        float4 f = ((const float4*)s)[i];
        union { __bf16 h[4]; uint2 u; } p;
        p.h[0] = (__bf16)f.x; p.h[1] = (__bf16)f.y;
        p.h[2] = (__bf16)f.z; p.h[3] = (__bf16)f.w;
        ((uint2*)d)[i] = p.u;
    }
}

// ---------------------------------------------------------------------------
// Fused GEMM:  Y[b][co][n] = sum_ci W[co][ci] * X[b][ci][n]
// plus per-channel sum / sumsq for BN (statStride=0) or IN (statStride=2*Co).
// Block: 128 threads (4 waves); 64 voxels per block; all Co, all Ci.
//  - B operand (wave's 16 voxels x Ci) cached in registers (Ci/32 x 8 VGPR).
//  - A operand fetched as clauses of ds_load_b128 (chunked), overlapping WMMA.
//  - Stats accumulated with no-return ds_add_f32 into an LDS array (no
//    cross-lane shuffles, no DS waits in the hot loop), flushed once per
//    block with global atomics.
// ---------------------------------------------------------------------------
template <int Ci, int Co>
__global__ __launch_bounds__(128) void gemm_wmma_kernel(
    const __bf16* __restrict__ X,    // [B][Ci][NS] bf16
    const __bf16* __restrict__ Wm,   // [Co][Ci] bf16
    float*        __restrict__ Y,    // [B][Co][NS] f32
    float*        __restrict__ stats,// sum[Co], sumsq[Co] (+ b*statStride)
    int NS, int statStride)
{
    constexpr int KT    = Ci / 32;               // k tiles
    constexpr int CHUNK = (KT > 6) ? 6 : KT;     // A-tile register chunk
    extern __shared__ __align__(16) char smem[];
    __bf16* sW    = (__bf16*)smem;               // [Co][Ci]
    __bf16* sX    = sW + (size_t)Co * Ci;        // [64][Ci] transposed tile
    float*  sStat = (float*)(sX + (size_t)64 * Ci); // [2*Co]

    const int tid = threadIdx.x;
    const int b   = blockIdx.z;
    const int n0  = blockIdx.x * 64;

    X += (size_t)b * Ci * NS;
    Y += (size_t)b * Co * NS;
    float* sb = stats + (size_t)b * statStride;

    // Zero LDS stats.
    for (int i = tid; i < 2 * Co; i += 128) sStat[i] = 0.f;

    // Stage full weight matrix into LDS with b128 copies.
    {
        const int4* src = (const int4*)Wm;
        int4*       dst = (int4*)sW;
        constexpr int cnt = (Co * Ci) >> 3;
        for (int i = tid; i < cnt; i += 128) dst[i] = src[i];
    }
    // Stage 64-voxel tile transposed to [n][ci]: read two channels with b128
    // global loads, pack channel pairs, write b32 LDS stores.
    {
        constexpr int cnt = (Ci / 2) * 8;
        for (int i = tid; i < cnt; i += 128) {
            const int ci = (i >> 3) * 2;
            const int n8 = (i & 7) << 3;
            int4 da = *(const int4*)(X + (size_t)ci * NS + n0 + n8);
            int4 db = *(const int4*)(X + (size_t)(ci + 1) * NS + n0 + n8);
            const __bf16* ea = (const __bf16*)&da;
            const __bf16* eb = (const __bf16*)&db;
#pragma unroll
            for (int j = 0; j < 8; ++j) {
                union { __bf16 h[2]; unsigned int u; } p;
                p.h[0] = ea[j]; p.h[1] = eb[j];
                *(unsigned int*)(sX + (size_t)(n8 + j) * Ci + ci) = p.u;
            }
        }
    }
    __syncthreads();

    const int lane = tid & 31;
    const int ln   = lane & 15;
    const int hi   = lane >> 4;            // half-wave select
    const int nloc = (tid >> 5) * 16 + ln; // local voxel id 0..63
    const int gn   = n0 + nloc;            // global voxel id (within batch)

    // Cache the full B strip for this wave's 16 voxels in registers.
    // B (32x16), this code's order (A same): VGPR v holds K pair {2v,2v+1}; lanes>=16 add +16
    // -> dwords [hi*8 .. hi*8+7] of each 32-element k tile: 2 x b128.
    U16BF Br[KT];
    {
        const int4* rowB = (const int4*)(sX + (size_t)nloc * Ci);
#pragma unroll
        for (int kt = 0; kt < KT; ++kt) {
            Br[kt].q[0] = rowB[kt * 4 + hi * 2];
            Br[kt].q[1] = rowB[kt * 4 + hi * 2 + 1];
        }
    }

    for (int co = 0; co < Co; co += 16) {
        // A (16x32), this code's order (matches B): VGPR v: K = 2v + 16*hi
        // -> dwords [hi*8 .. hi*8+7] of each 32-element k tile: 2 x b128.
        const int4* rowA = (const int4*)(sW + (size_t)(co + ln) * Ci);
        v8f acc = {};
#pragma unroll
        for (int kc = 0; kc < KT; kc += CHUNK) {
            U16BF Aall[CHUNK];
#pragma unroll
            for (int j = 0; j < CHUNK; ++j) {    // one clause of b128 loads
                Aall[j].q[0] = rowA[(kc + j) * 4 + hi * 2];
                Aall[j].q[1] = rowA[(kc + j) * 4 + hi * 2 + 1];
            }
#pragma unroll
            for (int j = 0; j < CHUNK; ++j)      // WMMAs overlap later loads
                acc = __builtin_amdgcn_wmma_f32_16x16x32_bf16(
                    false, Aall[j].v, false, Br[kc + j].v, (short)0, acc,
                    false, false);
        }
        // Epilogue: store + no-return LDS atomic stats (no waits, no shuffles)
#pragma unroll
        for (int v = 0; v < 8; ++v) {
            const int cov = co + v + hi * 8;     // C/D layout: M = v + 8*hi
            const float val = acc[v];
            Y[(size_t)cov * NS + gn] = val;
            atomicAdd(&sStat[cov], val);
            atomicAdd(&sStat[Co + cov], val * val);
        }
    }
    __syncthreads();
    // Flush block-level stats to global.
    for (int i = tid; i < 2 * Co; i += 128) atomicAdd(&sb[i], sStat[i]);
}

__device__ __forceinline__ void bn_coef(const float* sb, const float* g,
                                        const float* be, int c, int Co,
                                        float count, float& sc, float& sh)
{
    float mean = sb[c] / count;
    float var  = sb[Co + c] / count - mean * mean;
    float rs   = rsqrtf(var + EPSV);
    sc = g[c] * rs;
    sh = be[c] - mean * sc;
}

// ---------------------------------------------------------------------------
// Residue-class mins along W and H for one (b,c,d) plane (BN applied on the
// fly).  minW: [B][96][32][64][2]   minH: [B][96][32][2][64]
// ---------------------------------------------------------------------------
__global__ __launch_bounds__(256) void minwh_kernel(
    const float* __restrict__ Y, const float* __restrict__ stats,
    const float* __restrict__ g, const float* __restrict__ be, float count,
    float* __restrict__ minW, float* __restrict__ minH)
{
    __shared__ float sp[4096];
    const int d = blockIdx.x, c = blockIdx.y, b = blockIdx.z;
    float sc, sh; bn_coef(stats, g, be, c, 96, count, sc, sh);

    const float* Yp = Y + (((size_t)b * 96 + c) * 32 + d) * 4096;
    for (int i = threadIdx.x; i < 4096; i += 256) sp[i] = sc * Yp[i] + sh;
    __syncthreads();

    const size_t obase = (((size_t)b * 96 + c) * 32 + d) * 128;
    const int t = threadIdx.x;
    if (t < 128) {                       // minW: per (h, w%2), min over 32 w's
        const int h = t >> 1, par = t & 1;
        float m = 3.4e38f;
        for (int w = par; w < 64; w += 2) m = fminf(m, sp[h * 64 + w]);
        minW[obase + h * 2 + par] = m;
    } else {                             // minH: per (h%2, w), min over 32 h's
        const int u = t - 128, w = u >> 1, par = u & 1;
        float m = 3.4e38f;
        for (int h = par; h < 64; h += 2) m = fminf(m, sp[h * 64 + w]);
        minH[obase + par * 64 + w] = m;
    }
}

// minD: [B][96][2][64*64] — min over the 16 depth slices of each parity.
__global__ __launch_bounds__(256) void mind_kernel(
    const float* __restrict__ Y, const float* __restrict__ stats,
    const float* __restrict__ g, const float* __restrict__ be, float count,
    float* __restrict__ minD)
{
    const int n = blockIdx.x * 256 + threadIdx.x;   // 0..4095 (h*64+w)
    const int c = blockIdx.y, b = blockIdx.z;
    float sc, sh; bn_coef(stats, g, be, c, 96, count, sc, sh);
    const float* Yp = Y + (((size_t)b * 96 + c) * 32) * 4096;
    for (int par = 0; par < 2; ++par) {
        float m = 3.4e38f;
        for (int d = par; d < 32; d += 2)
            m = fminf(m, sc * Yp[(size_t)d * 4096 + n] + sh);
        minD[(((size_t)b * 96 + c) * 2 + par) * 4096 + n] = m;
    }
}

// MRConv: write hc[0:96]=h (bf16) and hc[96:192]=h-min3 (bf16) in one pass.
__global__ __launch_bounds__(256) void mr_kernel(
    const float* __restrict__ Y, const float* __restrict__ stats,
    const float* __restrict__ g, const float* __restrict__ be, float count,
    const float* __restrict__ minW, const float* __restrict__ minH,
    const float* __restrict__ minD, __bf16* __restrict__ hc, int NS)
{
    const int n  = blockIdx.x * 256 + threadIdx.x;  // 0..4095
    const int d  = blockIdx.y;
    const int bc = blockIdx.z;                      // b*96 + c
    const int b  = bc / 96, c = bc % 96;
    float sc, sh; bn_coef(stats, g, be, c, 96, count, sc, sh);

    const int h = n >> 6, w = n & 63;
    const size_t bcd = (size_t)bc * 32 + d;
    const float val = sc * Y[bcd * 4096 + n] + sh;
    const float m = fminf(minW[bcd * 128 + h * 2 + (w & 1)],
                    fminf(minH[bcd * 128 + (h & 1) * 64 + w],
                          minD[((size_t)bc * 2 + (d & 1)) * 4096 + n]));
    const size_t nIdx = (size_t)d * 4096 + n;
    __bf16* base = hc + (size_t)b * 192 * NS;
    base[(size_t)c * NS + nIdx]        = (__bf16)val;
    base[(size_t)(96 + c) * NS + nIdx] = (__bf16)(val - m);
}

// ---------------------------------------------------------------------------
// Generic normalize: val = BN/IN(y) [, lrelu][, +res], write f32 and/or bf16.
// 4 voxels per thread (b128 f32 traffic, b64 bf16 stores).
// ---------------------------------------------------------------------------
__global__ __launch_bounds__(256) void norm_kernel(
    const float* __restrict__ Y, const float* __restrict__ stats,
    int statStride, const float* __restrict__ g, const float* __restrict__ be,
    float count, int lrelu, const float* __restrict__ res,
    float* __restrict__ outF, __bf16* __restrict__ outB, int Co, int NS)
{
    const int n4 = (blockIdx.x * 256 + threadIdx.x) * 4;
    const int c = blockIdx.y, b = blockIdx.z;
    const float* sb = stats + (size_t)b * statStride;
    float sc, sh; bn_coef(sb, g, be, c, Co, count, sc, sh);
    const size_t idx = ((size_t)b * Co + c) * NS + n4;

    float4 y = *(const float4*)(Y + idx);
    float vv[4] = {y.x, y.y, y.z, y.w};
#pragma unroll
    for (int j = 0; j < 4; ++j) {
        float v = vv[j] * sc + sh;
        if (lrelu) v = (v >= 0.f) ? v : SLOPEV * v;
        vv[j] = v;
    }
    if (res) {
        float4 r = *(const float4*)(res + idx);
        vv[0] += r.x; vv[1] += r.y; vv[2] += r.z; vv[3] += r.w;
    }
    if (outF) *(float4*)(outF + idx) = make_float4(vv[0], vv[1], vv[2], vv[3]);
    if (outB) {
        union { __bf16 h[4]; uint2 u; } p;
#pragma unroll
        for (int j = 0; j < 4; ++j) p.h[j] = (__bf16)vv[j];
        *(uint2*)(outB + idx) = p.u;
    }
}

// ---------------------------------------------------------------------------
// Host orchestration
// ---------------------------------------------------------------------------
extern "C" void kernel_launch(void* const* d_in, const int* in_sizes, int n_in,
                              void* d_out, int out_size, void* d_ws, size_t ws_size,
                              hipStream_t stream)
{
    (void)in_sizes; (void)n_in; (void)out_size; (void)ws_size;
    const int Bn = 2;
    const int NS = 32 * 64 * 64;                 // 131072 voxels per batch
    const size_t NT = (size_t)Bn * NS;

    const float* x      = (const float*)d_in[0];
    const float* w_fc1  = (const float*)d_in[1];
    const float* g_bn1  = (const float*)d_in[3];
    const float* be_bn1 = (const float*)d_in[4];
    const float* w_g    = (const float*)d_in[5];
    const float* g_in   = (const float*)d_in[7];
    const float* be_in  = (const float*)d_in[8];
    const float* w_fc2  = (const float*)d_in[9];
    const float* g_bn2  = (const float*)d_in[11];
    const float* be_bn2 = (const float*)d_in[12];
    const float* w_f1   = (const float*)d_in[13];
    const float* g_bnf1 = (const float*)d_in[15];
    const float* be_bnf1= (const float*)d_in[16];
    const float* w_f2   = (const float*)d_in[17];
    const float* g_bnf2 = (const float*)d_in[19];
    const float* be_bnf2= (const float*)d_in[20];

    char* ws = (char*)d_ws;
    size_t off = 0;
    auto take = [&](size_t bytes) {
        size_t o = off;
        off = (off + bytes + 255) & ~(size_t)255;
        return o;
    };

    // stats: [BN1: 192][IN: 2*384][BN2: 192][BNF1: 768][BNF2: 192] floats
    float* statsAll = (float*)(ws + take(2112 * sizeof(float)));
    float* s1  = statsAll;            // 2*96
    float* sIN = statsAll + 192;      // Bn * 2*192
    float* s2  = statsAll + 960;      // 2*96
    float* sF1 = statsAll + 1152;     // 2*384
    float* sF2 = statsAll + 1920;     // 2*96

    __bf16* w1b  = (__bf16*)(ws + take((size_t)96  * 96  * 2));
    __bf16* wgb  = (__bf16*)(ws + take((size_t)192 * 192 * 2));
    __bf16* w2b  = (__bf16*)(ws + take((size_t)96  * 192 * 2));
    __bf16* wf1b = (__bf16*)(ws + take((size_t)384 * 96  * 2));
    __bf16* wf2b = (__bf16*)(ws + take((size_t)96  * 384 * 2));

    float* minW = (float*)(ws + take((size_t)Bn * 96 * 32 * 128 * 4));
    float* minH = (float*)(ws + take((size_t)Bn * 96 * 32 * 128 * 4));
    float* minD = (float*)(ws + take((size_t)Bn * 96 * 2 * 4096 * 4));

    // Arenas with time-multiplexed reuse:
    __bf16* bufC = (__bf16*)(ws + take((size_t)Bn * 192 * NS * 2)); // xbf -> hg
    float*  bufD = (float*) (ws + take((size_t)Bn * 96  * NS * 4)); // y1->y3->y5
    __bf16* bufB = (__bf16*)(ws + take((size_t)Bn * 384 * NS * 2)); // hc -> h4
    float*  bufA = (float*) (ws + take((size_t)Bn * 384 * NS * 4)); // y2 -> y4
    float*  x2   = (float*) (ws + take((size_t)Bn * 96  * NS * 4)); // residual 2
    __bf16* x2b  = (__bf16*)(ws + take((size_t)Bn * 96  * NS * 2));

    __bf16* xbf = bufC;  __bf16* hg = bufC;
    float*  y1  = bufD;  float*  y3 = bufD;  float* y5 = bufD;
    __bf16* hc  = bufB;  __bf16* h4 = bufB;
    float*  y2  = bufA;  float*  y4 = bufA;

    hipMemsetAsync(statsAll, 0, 2112 * sizeof(float), stream);

    auto cvt = [&](const float* s, __bf16* d, size_t n) {
        const unsigned n4 = (unsigned)(n / 4);
        cvt_f32_bf16_kernel<<<dim3((n4 + 255) / 256), 256, 0, stream>>>(s, d, (int)n4);
    };
    cvt(w_fc1, w1b,  (size_t)96 * 96);
    cvt(w_g,   wgb,  (size_t)192 * 192);
    cvt(w_fc2, w2b,  (size_t)96 * 192);
    cvt(w_f1,  wf1b, (size_t)384 * 96);
    cvt(w_f2,  wf2b, (size_t)96 * 384);
    cvt(x,     xbf,  NT * 96);

    const dim3 gGrid(NS / 64, 1, Bn);
    const float cntBN = (float)(Bn * (size_t)NS);
    const float cntIN = (float)NS;
    auto smemFor = [](int Ci, int Co) {
        return (size_t)(Co * Ci + 64 * Ci) * 2 + (size_t)2 * Co * 4;
    };

    // ---- Grapher ----
    // G1: y1 = x @ w_fc1^T  (+BN1 stats; bias cancels in BN)
    gemm_wmma_kernel<96, 96><<<gGrid, 128, smemFor(96, 96), stream>>>(
        xbf, w1b, y1, s1, NS, 0);
    // MRConv residue mins on h = BN1(y1)
    minwh_kernel<<<dim3(32, 96, Bn), 256, 0, stream>>>(
        y1, s1, g_bn1, be_bn1, cntBN, minW, minH);
    mind_kernel<<<dim3(16, 96, Bn), 256, 0, stream>>>(
        y1, s1, g_bn1, be_bn1, cntBN, minD);
    // hc = concat[h, h - min3]  (bf16)
    mr_kernel<<<dim3(16, 32, Bn * 96), 256, 0, stream>>>(
        y1, s1, g_bn1, be_bn1, cntBN, minW, minH, minD, hc, NS);
    // G2: y2 = hc @ w_g^T  (+IN stats per batch; bias cancels in IN)
    gemm_wmma_kernel<192, 192><<<gGrid, 128, smemFor(192, 192), stream>>>(
        hc, wgb, y2, sIN, NS, 2 * 192);
    // hg = lrelu(IN(y2))  (bf16)
    norm_kernel<<<dim3(NS / 1024, 192, Bn), 256, 0, stream>>>(
        y2, sIN, 2 * 192, g_in, be_in, cntIN, 1, nullptr, nullptr, hg, 192, NS);
    // G3: y3 = hg @ w_fc2^T  (+BN2 stats)
    gemm_wmma_kernel<192, 96><<<gGrid, 128, smemFor(192, 96), stream>>>(
        hg, w2b, y3, s2, NS, 0);
    // x2 = BN2(y3) + x  (f32 residual + bf16 copy for FFN)
    norm_kernel<<<dim3(NS / 1024, 96, Bn), 256, 0, stream>>>(
        y3, s2, 0, g_bn2, be_bn2, cntBN, 0, x, x2, x2b, 96, NS);

    // ---- FFN ----
    gemm_wmma_kernel<96, 384><<<gGrid, 128, smemFor(96, 384), stream>>>(
        x2b, wf1b, y4, sF1, NS, 0);
    norm_kernel<<<dim3(NS / 1024, 384, Bn), 256, 0, stream>>>(
        y4, sF1, 0, g_bnf1, be_bnf1, cntBN, 1, nullptr, nullptr, h4, 384, NS);
    gemm_wmma_kernel<384, 96><<<gGrid, 128, smemFor(384, 96), stream>>>(
        h4, wf2b, y5, sF2, NS, 0);
    norm_kernel<<<dim3(NS / 1024, 96, Bn), 256, 0, stream>>>(
        y5, sF2, 0, g_bnf2, be_bnf2, cntBN, 0, x2, (float*)d_out, nullptr, 96, NS);
}
